// ResampledGaussian_25898652795617
// MI455X (gfx1250) — matrix operand order repaired and checked by reading, not compiled, not run on hardware
//
#include <hip/hip_runtime.h>
#include <math.h>

typedef _Float16 v16h __attribute__((ext_vector_type(16)));
typedef _Float16 v8h  __attribute__((ext_vector_type(8)));
typedef _Float16 v2h  __attribute__((ext_vector_type(2)));
typedef float    v8f  __attribute__((ext_vector_type(8)));

#define LDS_STRIDE 72  // 64 K-values + 8 halfs pad; 144B row -> 16B-aligned b128 LDS loads

// ---------------------------------------------------------------------------
// Kernel 0: precompute loc/inv_scale pairs and sum(log_scale)
// ---------------------------------------------------------------------------
__global__ __launch_bounds__(64) void prep_kernel(const float* __restrict__ loc,
                                                  const float* __restrict__ log_scale,
                                                  float* __restrict__ hdr,
                                                  float* __restrict__ pre) {
  int j = threadIdx.x;
  if (j < 64) {
    pre[2 * j]     = loc[j];
    pre[2 * j + 1] = __expf(-log_scale[j]);
  }
  if (j == 0) {
    float s = 0.f;
    for (int k = 0; k < 64; ++k) s += log_scale[k];
    hdr[0] = s;  // sum of log_scale
  }
}

// ---------------------------------------------------------------------------
// Kernel 1: fused  relu(z@W1+b1)@W2+b2 -> sigmoid  +  gaussian log-density.
// One wave per 16 rows of z. WMMA f32_16x16x32_f16, K=64 in 2 steps.
// ---------------------------------------------------------------------------
__global__ __launch_bounds__(256) void fused_kernel(
    const float* __restrict__ z, const float* __restrict__ W1,
    const float* __restrict__ b1, const float* __restrict__ W2,
    const float* __restrict__ b2, const float* __restrict__ hdr,
    const float* __restrict__ pre, float* __restrict__ accbuf,
    float* __restrict__ lpg_out, float* __restrict__ partials, int nchunks) {
  __shared__ _Float16 w1t[256 * LDS_STRIDE];  // W1^T in f16, col-major [n][k]
  __shared__ float s_part[8];
  const int tid = threadIdx.x;

  // Stage W1 (row-major [k][n] f32) -> LDS transposed f16 [n][k].
  // Consecutive lanes read consecutive n (coalesced); pack 2 k's per b32 store.
  for (int t = tid; t < 8192; t += 256) {
    int n = t & 255;
    int k = (t >> 8) << 1;
    float a0 = W1[(size_t)k * 256 + n];
    float a1 = W1[(size_t)(k + 1) * 256 + n];
    v2h p; p[0] = (_Float16)a0; p[1] = (_Float16)a1;
    *(v2h*)&w1t[n * LDS_STRIDE + k] = p;
  }
  __syncthreads();

  const int wave  = tid >> 5;
  const int lane  = tid & 31;
  const int hs    = lane >> 4;        // half-wave select
  const int chunk = blockIdx.x * 8 + wave;
  float wave_part = 0.f;

  if (chunk < nchunks) {              // wave-uniform guard: EXEC all-1 inside
    const int base = chunk << 4;
    const int row  = base + (lane & 15);
    const float4* zq = (const float4*)(z + (size_t)row * 64);

    // ---- Build A fragments (ISA 16-bit A 16x32 layout) + gaussian partial --
    float q = 0.f;
    v16h afr[2];
#pragma unroll
    for (int kt = 0; kt < 2; ++kt) {
      const int j0 = kt * 32 + hs * 8;        // first 8-run of K
      const int j1 = kt * 32 + 16 + hs * 8;   // second 8-run of K
      float4 za = zq[j0 >> 2], zb = zq[(j0 >> 2) + 1];
      float4 zc = zq[j1 >> 2], zd = zq[(j1 >> 2) + 1];
      float zf[16] = {za.x, za.y, za.z, za.w, zb.x, zb.y, zb.z, zb.w,
                      zc.x, zc.y, zc.z, zc.w, zd.x, zd.y, zd.z, zd.w};
      v16h a;
#pragma unroll
      for (int i = 0; i < 8; ++i) {
        float lc = pre[2 * (j0 + i)], iv = pre[2 * (j0 + i) + 1];
        float t = (zf[i] - lc) * iv;
        q = fmaf(t, t, q);
        a[i] = (_Float16)zf[i];
      }
#pragma unroll
      for (int i = 0; i < 8; ++i) {
        float lc = pre[2 * (j1 + i)], iv = pre[2 * (j1 + i) + 1];
        float t = (zf[8 + i] - lc) * iv;
        q = fmaf(t, t, q);
        a[8 + i] = (_Float16)zf[8 + i];
      }
      afr[kt] = a;
    }

    // ---- 16 column tiles: WMMA + fused relu/bias/W2 epilogue ---------------
    float accr[8];
#pragma unroll
    for (int r = 0; r < 8; ++r) accr[r] = 0.f;

    const int ncl = lane & 15;
#pragma unroll
    for (int ct = 0; ct < 16; ++ct) {
      const int ncol = ct * 16 + ncl;
      const int boff = ncol * LDS_STRIDE + (hs << 3);  // lanes<16: K0..7,16..23; lanes>=16: K8..15,24..31
      v8h lo0 = *(const v8h*)&w1t[boff];
      v8h hi0 = *(const v8h*)&w1t[boff + 16];
      v8h lo1 = *(const v8h*)&w1t[boff + 32];
      v8h hi1 = *(const v8h*)&w1t[boff + 48];
      v16h bf0, bf1;
#pragma unroll
      for (int i = 0; i < 8; ++i) {
        bf0[i] = lo0[i]; bf0[8 + i] = hi0[i];
        bf1[i] = lo1[i]; bf1[8 + i] = hi1[i];
      }
      v8f c = {};
      c = __builtin_amdgcn_wmma_f32_16x16x32_f16(false, afr[0], false, bf0,
                                                 (short)0, c, false, false);
      c = __builtin_amdgcn_wmma_f32_16x16x32_f16(false, afr[1], false, bf1,
                                                 (short)0, c, false, false);
      float biasv = b1[ncol];
      float w2v   = W2[ncol];
#pragma unroll
      for (int r = 0; r < 8; ++r) {
        float h = fmaxf(c[r] + biasv, 0.f);   // relu(hidden)
        accr[r] = fmaf(h, w2v, accr[r]);      // dot with W2
      }
    }

    // Butterfly-reduce the 16 columns held per half-wave (row sums everywhere)
#pragma unroll
    for (int r = 0; r < 8; ++r) {
      float v = accr[r];
      v += __shfl_xor(v, 1);
      v += __shfl_xor(v, 2);
      v += __shfl_xor(v, 4);
      v += __shfl_xor(v, 8);
      accr[r] = v;
    }

    // Gaussian log-density: lane L and L+16 hold complementary halves of row
    float qt  = q + __shfl_xor(q, 16);
    float lpg = -58.81206612510958f /* -0.5*64*log(2*pi) */ - hdr[0] - 0.5f * qt;
    if (lane < 16) lpg_out[row] = lpg;

    // Sigmoid + store acceptance; lanes 0 and 16 own rows 0..7 / 8..15
    float local = 0.f;
    float b2v = b2[0];
    if ((lane & 15) == 0) {
#pragma unroll
      for (int r = 0; r < 8; ++r) {
        float logit = accr[r] + b2v;
        float s = 1.f / (1.f + __expf(-logit));
        accbuf[base + hs * 8 + r] = s;
        local += s;
      }
    }
    wave_part = local + __shfl_xor(local, 16);
  }

  if (lane == 0) s_part[wave] = wave_part;
  __syncthreads();
  if (tid == 0) {
    float t = 0.f;
#pragma unroll
    for (int i = 0; i < 8; ++i) t += s_part[i];
    partials[blockIdx.x] = t;  // deterministic per-block partial
  }
}

// ---------------------------------------------------------------------------
// Kernel 2: deterministic reduction of partials -> Z, alpha, (1-alpha)/Z
// ---------------------------------------------------------------------------
__global__ __launch_bounds__(256) void reduce_kernel(const float* __restrict__ partials,
                                                     int nblk, float* __restrict__ hdr,
                                                     float invN) {
  __shared__ float s[256];
  float t = 0.f;
  for (int i = threadIdx.x; i < nblk; i += 256) t += partials[i];
  s[threadIdx.x] = t;
  __syncthreads();
  for (int st = 128; st > 0; st >>= 1) {
    if (threadIdx.x < st) s[threadIdx.x] += s[threadIdx.x + st];
    __syncthreads();
  }
  if (threadIdx.x == 0) {
    float Z = s[0] * invN;
    float alpha = __powf(1.f - Z, 99.f);  // (1-Z)^(T-1), T=100
    hdr[1] = Z;
    hdr[2] = (1.f - alpha) / Z;
    hdr[3] = alpha;
  }
}

// ---------------------------------------------------------------------------
// Kernel 3: log_p = log(coefA*acc + alpha) + log_p_gauss (in d_out)
// ---------------------------------------------------------------------------
__global__ __launch_bounds__(256) void finalize_kernel(const float* __restrict__ accbuf,
                                                       const float* __restrict__ hdr,
                                                       float* __restrict__ out, int n) {
  int i = blockIdx.x * blockDim.x + threadIdx.x;
  if (i < n) {
    float coefA = hdr[2];
    float alpha = hdr[3];
    out[i] = logf(fmaf(coefA, accbuf[i], alpha)) + out[i];
  }
}

// ---------------------------------------------------------------------------
extern "C" void kernel_launch(void* const* d_in, const int* in_sizes, int n_in,
                              void* d_out, int out_size, void* d_ws, size_t ws_size,
                              hipStream_t stream) {
  const float* z   = (const float*)d_in[0];
  const float* loc = (const float*)d_in[1];
  const float* lsc = (const float*)d_in[2];
  const float* W1  = (const float*)d_in[3];
  const float* b1  = (const float*)d_in[4];
  const float* W2  = (const float*)d_in[5];
  const float* b2  = (const float*)d_in[6];
  float* out = (float*)d_out;

  const int N = out_size;               // 500000 rows
  const int nchunks = (N + 15) / 16;    // 16 rows per wave
  const int blocks1 = (nchunks + 7) / 8;  // 8 waves per 256-thread block

  // workspace layout (floats):
  // [0..7]   header: sls, Z, (1-alpha)/Z, alpha
  // [8..135] pre: 64 x (loc, inv_scale) pairs
  // [136..]  per-block partials
  // [8192..] acceptance buffer [N]
  float* wsf      = (float*)d_ws;
  float* hdr      = wsf;
  float* pre      = wsf + 8;
  float* partials = wsf + 136;
  float* accbuf   = wsf + 8192;

  prep_kernel<<<1, 64, 0, stream>>>(loc, lsc, hdr, pre);

  fused_kernel<<<blocks1, 256, 0, stream>>>(z, W1, b1, W2, b2, hdr, pre,
                                            accbuf, out, partials, nchunks);

  reduce_kernel<<<1, 256, 0, stream>>>(partials, blocks1, hdr, 1.0f / (float)N);

  finalize_kernel<<<(N + 255) / 256, 256, 0, stream>>>(accbuf, hdr, out, N);
}
